// GatedBlock_89584427860071
// MI455X (gfx1250) — matrix operand order repaired and checked by reading, not compiled, not run on hardware
//
#include <hip/hip_runtime.h>
#include <hip/hip_bf16.h>

typedef __attribute__((ext_vector_type(16))) __bf16 v16bf;
typedef __attribute__((ext_vector_type(8)))  float  v8f;
typedef __attribute__((ext_vector_type(4)))  unsigned int v4u;
typedef __attribute__((ext_vector_type(8)))  int  v8i;
typedef __attribute__((ext_vector_type(4)))  int  v4i;

#define NROWS       32768
#define K_IN        2304
#define N_FIELDS    2304
#define N_GATES     512
#define N_TOTAL     2816

#define BM 128
#define BN 256
#define BK 32
#define LDS_STRIDE (BK + 8)   // 40 bf16 = 80B LDS rows (64B data + 16B TDM pad)

#if __has_builtin(__builtin_amdgcn_tensor_load_to_lds) && __has_builtin(__builtin_amdgcn_s_wait_tensorcnt)
#define USE_TDM 1
#else
#define USE_TDM 0
#endif

__device__ __forceinline__ float fast_sigmoid(float f) {
    return __builtin_amdgcn_rcpf(1.0f + __expf(-f));   // v_exp_f32 + v_rcp_f32
}

// ---------------- conversion kernels ----------------

__global__ __launch_bounds__(256) void convert_x_kernel(const float* __restrict__ x,
                                                        __bf16* __restrict__ xb) {
    long i = ((long)blockIdx.x * 256 + threadIdx.x) * 4;
    float4 v = *(const float4*)(x + i);
    union { __bf16 b[4]; unsigned long long u; } p;
    p.b[0] = (__bf16)v.x; p.b[1] = (__bf16)v.y;
    p.b[2] = (__bf16)v.z; p.b[3] = (__bf16)v.w;
    *(unsigned long long*)(xb + i) = p.u;
}

// Wt[n][k] = W[k][n]   (W is K_IN x N_TOTAL row-major)
__global__ __launch_bounds__(256) void convert_wt_kernel(const float* __restrict__ W,
                                                         __bf16* __restrict__ wt) {
    int k = blockIdx.x * 256 + threadIdx.x;
    int n = blockIdx.y;
    wt[(size_t)n * K_IN + k] = (__bf16)W[(size_t)k * N_TOTAL + n];
}

#if USE_TDM
// Issue one 2D TDM tile load: rows x 32 bf16, row pitch K_IN elements,
// LDS rows padded 64B->80B by TDM (pad_interval=3 => 64B, pad_amount=3 => 16B).
__device__ __forceinline__ void tdm_tile_load(const __bf16* gsrc, unsigned lds_addr,
                                              int rows) {
    unsigned long long ga = (unsigned long long)(uintptr_t)gsrc;
    v4u g0;
    g0.x = 1u;                                       // count=1, user mode
    g0.y = lds_addr;                                 // LDS byte address
    g0.z = (unsigned)(ga & 0xffffffffull);           // global_addr[31:0]
    g0.w = (unsigned)((ga >> 32) & 0x1ffffffull)     // global_addr[56:32]
         | (2u << 30);                               // type = 2 (image)
    v8i g1;
    g1[0] = (int)((1u << 16)                         // data_size = 2 bytes
                | (1u << 20)                         // pad_enable
                | (3u << 22)                         // pad_interval: 64B
                | (3u << 25));                       // pad_amount: 16B
    g1[1] = (int)((BK & 0xffff) << 16);              // tensor_dim0 = 32 (lo16)
    g1[2] = (int)((rows & 0xffff) << 16);            // dim0 hi=0 | tensor_dim1 lo16
    g1[3] = (int)((BK & 0xffff) << 16);              // dim1 hi=0 | tile_dim0 = 32
    g1[4] = (int)(rows & 0xffff);                    // tile_dim1 = rows, tile_dim2=0
    g1[5] = (int)K_IN;                               // tensor_dim0_stride lo32
    g1[6] = 0;                                       // stride hi | dim1_stride lo
    g1[7] = 0;
    v4i gz4 = {0, 0, 0, 0};
    v8i gz8 = {0, 0, 0, 0, 0, 0, 0, 0};
    __builtin_amdgcn_tensor_load_to_lds(g0, g1, gz4, gz4, gz8, 0);
}
#endif

// ---------------- GEMM + epilogue ----------------
// Block 128x256, 8 waves as 2(M) x 4(N), wave tile 64x64 = 16 WMMA accs.
// Double-buffered LDS; tiles delivered by the Tensor Data Mover (TENSORcnt).

template<bool GATE_PASS>
__global__ __launch_bounds__(256, 2) void gemm_gated_kernel(
    const __bf16* __restrict__ xb,     // [NROWS][K_IN]
    const __bf16* __restrict__ wt,     // [pass-N][K_IN]
    const float*  __restrict__ gates,  // [NROWS][N_GATES] (main pass only)
    float*        __restrict__ out)
{
    __shared__ __align__(16) __bf16 As[2][BM * LDS_STRIDE];
    __shared__ __align__(16) __bf16 Bs[2][BN * LDS_STRIDE];

    const int t      = threadIdx.x;
    const int lane   = t & 31;
    const int laneLo = lane & 15;
    const int laneHi = lane >> 4;
    const int wave   = t >> 5;
    const int wm     = wave >> 2;   // 0..1 : 64 rows
    const int wn     = wave & 3;    // 0..3 : 64 cols

    const int m0 = blockIdx.x * BM;
    const int n0 = blockIdx.y * BN;

    const __bf16* aTileBase = xb + (size_t)m0 * K_IN;
    const __bf16* bTileBase = wt + (size_t)n0 * K_IN;

    v8f acc[4][4];
#pragma unroll
    for (int mt = 0; mt < 4; ++mt)
#pragma unroll
        for (int nt = 0; nt < 4; ++nt)
            acc[mt][nt] = (v8f){0.f, 0.f, 0.f, 0.f, 0.f, 0.f, 0.f, 0.f};

#if USE_TDM
    const unsigned ldsA[2] = {(unsigned)(uintptr_t)&As[0][0], (unsigned)(uintptr_t)&As[1][0]};
    const unsigned ldsB[2] = {(unsigned)(uintptr_t)&Bs[0][0], (unsigned)(uintptr_t)&Bs[1][0]};

    if (wave == 0) {
        tdm_tile_load(aTileBase, ldsA[0], BM);
        tdm_tile_load(bTileBase, ldsB[0], BN);
        __builtin_amdgcn_s_wait_tensorcnt(0);
    }
    __syncthreads();
#else
    const int arow0 = t >> 2;
    const int kc0   = (t & 3) * 8;
    uint4 areg[2], breg[4];
#pragma unroll
    for (int i = 0; i < 2; ++i)
        areg[i] = *(const uint4*)(aTileBase + (size_t)(arow0 + i * 64) * K_IN + kc0);
#pragma unroll
    for (int i = 0; i < 4; ++i)
        breg[i] = *(const uint4*)(bTileBase + (size_t)(arow0 + i * 64) * K_IN + kc0);
#pragma unroll
    for (int i = 0; i < 2; ++i)
        *(uint4*)(&As[0][(arow0 + i * 64) * LDS_STRIDE + kc0]) = areg[i];
#pragma unroll
    for (int i = 0; i < 4; ++i)
        *(uint4*)(&Bs[0][(arow0 + i * 64) * LDS_STRIDE + kc0]) = breg[i];
    __syncthreads();
#endif

    int cur = 0;
    for (int kb = 0; kb < K_IN; kb += BK) {
        const bool more = (kb + BK) < K_IN;

#if USE_TDM
        if (more && wave == 0) {
            tdm_tile_load(aTileBase + kb + BK, ldsA[cur ^ 1], BM);
            tdm_tile_load(bTileBase + kb + BK, ldsB[cur ^ 1], BN);
        }
#else
        if (more) {
#pragma unroll
            for (int i = 0; i < 2; ++i)
                areg[i] = *(const uint4*)(aTileBase + (size_t)(arow0 + i * 64) * K_IN + kb + BK + kc0);
#pragma unroll
            for (int i = 0; i < 4; ++i)
                breg[i] = *(const uint4*)(bTileBase + (size_t)(arow0 + i * 64) * K_IN + kb + BK + kc0);
        }
#endif

        // fragment loaders (A 16x32: K chunks [hi*8,hi*8+8) and +16;
        //                   B 32x16: same K chunks as A)
        auto loadA = [&](int mt) -> v16bf {
            const __bf16* p = &As[cur][(wm * 64 + mt * 16 + laneLo) * LDS_STRIDE
                                       + laneHi * 8];
            union { uint4 u[2]; v16bf v; } tmp;
            tmp.u[0] = *(const uint4*)(p);
            tmp.u[1] = *(const uint4*)(p + 16);
            return tmp.v;
        };
        auto loadB = [&](int nt) -> v16bf {
            const __bf16* p = &Bs[cur][(wn * 64 + nt * 16 + laneLo) * LDS_STRIDE
                                       + laneHi * 8];
            union { uint4 u[2]; v16bf v; } tmp;
            tmp.u[0] = *(const uint4*)(p);
            tmp.u[1] = *(const uint4*)(p + 16);
            return tmp.v;
        };

        v16bf bfrag[4];
#pragma unroll
        for (int nt = 0; nt < 4; ++nt) bfrag[nt] = loadB(nt);

        // 2-deep A-fragment pipeline with two distinct live registers, so the
        // ds_loads for group g+2 issue after group g's WMMAs consume a0/a1, before group g+1's.
        v16bf a0 = loadA(0);
        v16bf a1 = loadA(1);
#pragma unroll
        for (int nt = 0; nt < 4; ++nt)
            acc[0][nt] = __builtin_amdgcn_wmma_f32_16x16x32_bf16(
                false, a0, false, bfrag[nt], (short)0, acc[0][nt], false, false);
        a0 = loadA(2);
#pragma unroll
        for (int nt = 0; nt < 4; ++nt)
            acc[1][nt] = __builtin_amdgcn_wmma_f32_16x16x32_bf16(
                false, a1, false, bfrag[nt], (short)0, acc[1][nt], false, false);
        a1 = loadA(3);
#pragma unroll
        for (int nt = 0; nt < 4; ++nt)
            acc[2][nt] = __builtin_amdgcn_wmma_f32_16x16x32_bf16(
                false, a0, false, bfrag[nt], (short)0, acc[2][nt], false, false);
#pragma unroll
        for (int nt = 0; nt < 4; ++nt)
            acc[3][nt] = __builtin_amdgcn_wmma_f32_16x16x32_bf16(
                false, a1, false, bfrag[nt], (short)0, acc[3][nt], false, false);

#if USE_TDM
        if (more && wave == 0)
            __builtin_amdgcn_s_wait_tensorcnt(0);
#else
        if (more) {
#pragma unroll
            for (int i = 0; i < 2; ++i)
                *(uint4*)(&As[cur ^ 1][(arow0 + i * 64) * LDS_STRIDE + kc0]) = areg[i];
#pragma unroll
            for (int i = 0; i < 4; ++i)
                *(uint4*)(&Bs[cur ^ 1][(arow0 + i * 64) * LDS_STRIDE + kc0]) = breg[i];
        }
#endif
        __syncthreads();
        cur ^= 1;
    }

    // ---- epilogue: D layout M = r + hi*8, N = laneLo ----
#pragma unroll
    for (int mt = 0; mt < 4; ++mt) {
#pragma unroll
        for (int nt = 0; nt < 4; ++nt) {
#pragma unroll
            for (int r = 0; r < 8; ++r) {
                int m = m0 + wm * 64 + mt * 16 + laneHi * 8 + r;
                int n = n0 + wn * 64 + nt * 16 + laneLo;
                float f = acc[mt][nt][r];
                if (GATE_PASS) {
                    out[(size_t)m * N_GATES + n] = fast_sigmoid(f);
                } else {
                    float o;
                    if (n < 256) {
                        o = f * fast_sigmoid(f);              // silu
                    } else {
                        int g = (n < 1024) ? (n - 256) / 3
                                           : 256 + (n - 1024) / 5;
                        o = f * gates[(size_t)m * N_GATES + g];
                    }
                    out[(size_t)m * N_FIELDS + n] = o;
                }
            }
        }
    }
}

// ---------------- launcher ----------------

extern "C" void kernel_launch(void* const* d_in, const int* in_sizes, int n_in,
                              void* d_out, int out_size, void* d_ws, size_t ws_size,
                              hipStream_t stream) {
    const float* x = (const float*)d_in[0];
    const float* W = (const float*)d_in[1];
    float* out = (float*)d_out;

    char* ws = (char*)d_ws;
    __bf16* xb = (__bf16*)ws;                                   // 151.0 MB
    __bf16* wt = (__bf16*)(ws + (size_t)NROWS * K_IN * 2);      //  13.0 MB
    float* gates = (float*)(ws + (size_t)NROWS * K_IN * 2
                               + (size_t)N_TOTAL * K_IN * 2);   //  67.1 MB

    // 1) x -> bf16
    convert_x_kernel<<<(NROWS * (long)K_IN) / (4 * 256), 256, 0, stream>>>(x, xb);
    // 2) W -> bf16 transposed
    convert_wt_kernel<<<dim3(K_IN / 256, N_TOTAL), 256, 0, stream>>>(W, wt);
    // 3) gate pass: sigmoid(x @ W[:, 2304:2816]) -> ws
    gemm_gated_kernel<true><<<dim3(NROWS / BM, N_GATES / BN), 256, 0, stream>>>(
        xb, wt + (size_t)N_FIELDS * K_IN, nullptr, gates);
    // 4) field pass + fused gating -> d_out
    gemm_gated_kernel<false><<<dim3(NROWS / BM, N_FIELDS / BN), 256, 0, stream>>>(
        xb, wt, gates, out);
}
